// MTNet_38603166056508
// MI455X (gfx1250) — matrix operand order repaired and checked by reading, not compiled, not run on hardware
//
#include <hip/hip_runtime.h>
#include <hip/hip_bf16.h>
#include <stdint.h>

// ---------------------------------------------------------------------------
// MTNet scoring on gfx1250:
//   K0: poi_emb fp32 -> bf16 (RNE) into workspace          (bandwidth trivial)
//   K1: per-batch attention MLP + softmax -> pref row bf16 (fp32 math)
//   K2: scores = pref @ poi_emb^T via v_wmma_f32_16x16x32_bf16
// Store stream of the 819MB fp32 score matrix is the HBM floor (~35us at
// 23.3 TB/s); poi_emb (51MB bf16) + pref (1MB) live in the 192MB L2, so the
// GEMM operands are L2 hits and bf16 WMMA keeps compute at/below that floor.
// ---------------------------------------------------------------------------

typedef __attribute__((ext_vector_type(16))) __bf16 v16bf;
typedef __attribute__((ext_vector_type(8)))  __bf16 v8bf;
typedef __attribute__((ext_vector_type(8)))  float  v8f;

#define EDIM 256   // embed size (reference E)
#define HDIM 128   // E/2 hidden
#define MAXL 8     // >= L (reference L = 6)

static __device__ __forceinline__ unsigned short f32_to_bf16_rne(float f) {
    union { float f; unsigned int u; } v; v.f = f;
    unsigned int u = v.u;
    unsigned int r = u + 0x7fffu + ((u >> 16) & 1u);   // round-to-nearest-even
    if ((u & 0x7f800000u) == 0x7f800000u) r = u;       // keep inf/nan payload
    return (unsigned short)(r >> 16);
}

// ---------------- K0: fp32 -> bf16 bits (grid-stride) ----------------------
__global__ void cvt_f32_bf16(const float* __restrict__ in,
                             unsigned short* __restrict__ out, long long n) {
    long long i = (long long)blockIdx.x * blockDim.x + threadIdx.x;
    long long stride = (long long)gridDim.x * blockDim.x;
    for (; i < n; i += stride) out[i] = f32_to_bf16_rne(in[i]);
}

// ---------------- K1: attention MLP + softmax -> pref (bf16) ---------------
// One 256-thread block per batch element; tid == embedding index e.
__global__ __launch_bounds__(EDIM) void attn_pref(
    const int* __restrict__ user, const int* __restrict__ last_poi,
    const int* __restrict__ path_nodes, const unsigned char* __restrict__ path_mask,
    const int* __restrict__ num_nodes_p,
    const float* __restrict__ poi_emb, const float* __restrict__ tree_emb,
    const float* __restrict__ w1, const float* __restrict__ b1,
    const float* __restrict__ w2, const float* __restrict__ b2,
    unsigned short* __restrict__ pref_bf16,   // [B, EDIM] bf16 bits
    int L) {
    const int b   = blockIdx.x;
    const int tid = threadIdx.x;               // == e
    const int NN  = num_nodes_p[0];            // num_nodes scalar (63)

    __shared__ float sh_tp[MAXL * EDIM];       // path_pref rows
    __shared__ float sh_h [MAXL * EDIM];       // tanh(lp + tp)
    __shared__ float sh_red[MAXL][HDIM];       // MLP reduction buffer
    __shared__ float sh_w [MAXL];              // softmax weights

    const int   u  = user[b];
    const float lp = poi_emb[(long long)last_poi[b] * EDIM + tid];
    for (int l = 0; l < L; ++l) {
        const int   node = path_nodes[b * L + l];
        const float tp   = tree_emb[((long long)u * NN + node) * EDIM + tid];
        sh_tp[l * EDIM + tid] = tp;
        sh_h [l * EDIM + tid] = tanhf(lp + tp);
    }
    __syncthreads();

    // z_l[j] = b1[j] + sum_e h_l[e]*w1[e][j];  s_l[j] = relu(z)*w2[j]
    if (tid < HDIM) {
        float z[MAXL];
        for (int l = 0; l < L; ++l) z[l] = b1[tid];
        for (int e = 0; e < EDIM; ++e) {
            const float wv = w1[e * HDIM + tid];   // coalesced over tid
            for (int l = 0; l < L; ++l)
                z[l] = fmaf(sh_h[l * EDIM + e], wv, z[l]);
        }
        const float w2v = w2[tid];
        for (int l = 0; l < L; ++l) sh_red[l][tid] = fmaxf(z[l], 0.0f) * w2v;
    }
    __syncthreads();
    for (int stride = HDIM / 2; stride > 0; stride >>= 1) {
        if (tid < stride)
            for (int l = 0; l < L; ++l) sh_red[l][tid] += sh_red[l][tid + stride];
        __syncthreads();
    }

    if (tid == 0) {    // masked softmax over L (L<=8, scalar is fine)
        float attn[MAXL]; float mx = -1e30f;
        for (int l = 0; l < L; ++l) {
            float a = sh_red[l][0] + b2[0];
            a = path_mask[b * L + l] ? a : -1e9f;
            attn[l] = a; mx = fmaxf(mx, a);
        }
        float ssum = 0.0f;
        for (int l = 0; l < L; ++l) { attn[l] = __expf(attn[l] - mx); ssum += attn[l]; }
        const float inv = 1.0f / ssum;
        for (int l = 0; l < L; ++l) sh_w[l] = attn[l] * inv;
    }
    __syncthreads();

    float pref = 0.0f;
    for (int l = 0; l < L; ++l) pref = fmaf(sh_w[l], sh_tp[l * EDIM + tid], pref);
    pref_bf16[(long long)b * EDIM + tid] = f32_to_bf16_rne(pref);
}

// ---------------- K2: scores = pref @ poi_emb^T (bf16 WMMA) ----------------
// Block = 8 waves (256 thr). Each wave owns a 16(M) x 64(N) strip of C:
// A-fragment loaded once per K-step, reused across 4 WMMAs.
__global__ __launch_bounds__(256) void score_gemm(
    const unsigned short* __restrict__ A,    // [B, EDIM] bf16 bits (pref)
    const unsigned short* __restrict__ Bm,   // [N, EDIM] bf16 bits (poi_emb)
    float* __restrict__ out,                 // [B, N] fp32
    int N) {
    const int lane  = threadIdx.x & 31;
    const int wave  = threadIdx.x >> 5;
    const int m_base = blockIdx.y * 16;
    const long long n0 = ((long long)blockIdx.x * 8 + wave) * 64;
    if (n0 >= N) return;                     // wave-uniform exit

    const int rlane = lane & 15;
    const int sel   = lane >> 4;

    // A fragment per lane: row = rlane; K halves {sel*8..+7} and {16+sel*8..+7}
    const unsigned short* arow = A + (long long)(m_base + rlane) * EDIM;

    // B fragment per lane: poi row = tile_base + rlane; K halves {sel*8..+7} and {16+sel*8..+7}
    const unsigned short* brow[4];
    bool tile_ok[4];
#pragma unroll
    for (int t = 0; t < 4; ++t) {
        const long long nb = n0 + t * 16;
        const long long r  = nb + rlane;
        tile_ok[t] = (nb < N);                         // N%16==0 -> whole tiles
        brow[t] = Bm + (r < N ? r : 0) * EDIM;         // clamp OOB reads
    }

    v8f acc[4] = {};

#pragma unroll
    for (int k = 0; k < EDIM; k += 32) {
        union { v16bf v; v8bf h[2]; } af;
        af.h[0] = *(const v8bf*)(arow + k +      sel * 8);
        af.h[1] = *(const v8bf*)(arow + k + 16 + sel * 8);
#pragma unroll
        for (int t = 0; t < 4; ++t) {
            union { v16bf v; v8bf h[2]; } bfr;
            const unsigned short* bp = brow[t] + k + sel * 8;
            bfr.h[0] = *(const v8bf*)(bp);
            bfr.h[1] = *(const v8bf*)(bp + 16);
            acc[t] = __builtin_amdgcn_wmma_f32_16x16x32_bf16(
                false, af.v, false, bfr.v, (short)0, acc[t], false, false);
        }
    }

    // C/D layout: VGPR i -> (M = i + 8*sel, N = rlane)
#pragma unroll
    for (int t = 0; t < 4; ++t) {
        if (!tile_ok[t]) continue;                     // wave-uniform
        const long long col = n0 + t * 16 + rlane;
        if (col < N) {
#pragma unroll
            for (int i = 0; i < 8; ++i) {
                const int row = m_base + i + sel * 8;
                out[(long long)row * N + col] = acc[t][i];
            }
        }
    }
}

// ---------------------------------------------------------------------------
extern "C" void kernel_launch(void* const* d_in, const int* in_sizes, int n_in,
                              void* d_out, int out_size, void* d_ws, size_t ws_size,
                              hipStream_t stream) {
    const int*           user       = (const int*)d_in[0];
    const int*           last_poi   = (const int*)d_in[1];
    const int*           path_nodes = (const int*)d_in[2];
    const unsigned char* path_mask  = (const unsigned char*)d_in[3];  // bool
    const int*           num_nodes  = (const int*)d_in[4];
    const float*         poi_emb    = (const float*)d_in[5];
    const float*         tree_emb   = (const float*)d_in[6];
    const float*         w1         = (const float*)d_in[7];
    const float*         b1         = (const float*)d_in[8];
    const float*         w2         = (const float*)d_in[9];
    const float*         b2         = (const float*)d_in[10];

    const int Bsz = in_sizes[0];
    const int L   = in_sizes[2] / Bsz;
    const long long NP = (long long)in_sizes[5] / EDIM;

    // workspace: [pref bf16 (B*E)] [poi bf16 (N*E)]
    unsigned short* pref_bf = (unsigned short*)d_ws;
    const size_t a_bytes = ((size_t)Bsz * EDIM * sizeof(unsigned short) + 255) & ~(size_t)255;
    unsigned short* poi_bf = (unsigned short*)((char*)d_ws + a_bytes);

    const long long poi_n = NP * EDIM;
    cvt_f32_bf16<<<dim3(2048), dim3(256), 0, stream>>>(poi_emb, poi_bf, poi_n);

    attn_pref<<<dim3(Bsz), dim3(EDIM), 0, stream>>>(
        user, last_poi, path_nodes, path_mask, num_nodes,
        poi_emb, tree_emb, w1, b1, w2, b2, pref_bf, L);

    const long long n_groups = (NP + 63) / 64;            // 16x64 strips per M-tile
    dim3 grid((unsigned)((n_groups + 7) / 8), (unsigned)(Bsz / 16));
    score_gemm<<<grid, dim3(256), 0, stream>>>(pref_bf, poi_bf, (float*)d_out, (int)NP);
}
